// ScaledDotProductAttention_72610717106572
// MI455X (gfx1250) — compile-verified, matrix operand order checked by reading, not run on hardware
//
#include <hip/hip_runtime.h>

typedef _Float16 h16v __attribute__((ext_vector_type(16)));
typedef _Float16 h8v  __attribute__((ext_vector_type(8)));
typedef float    f8v  __attribute__((ext_vector_type(8)));
typedef float    f4v  __attribute__((ext_vector_type(4)));
typedef int      i4v  __attribute__((ext_vector_type(4)));

#define SEQ_L 2048
#define DH    64
#define BC    64            // keys per iteration
#define NW    8             // waves per workgroup
#define BR    (16*NW)       // 128 queries per workgroup
#define KST   72            // LDS row stride in halfs (pad vs. bank conflicts; keeps 16B align)
#define NBLK  (SEQ_L/BC)

union H16 { h16v v; h8v h[2]; _Float16 e[16]; };
union H8  { h8v  v; _Float16 e[8]; };

__global__ __launch_bounds__(256, 1)
void fattn_kernel(const float* __restrict__ Qg, const float* __restrict__ Kg,
                  const float* __restrict__ Vg, const int* __restrict__ Mg,
                  float* __restrict__ Og)
{
    // double-buffered K (row-major, key x d) and V^T (d x key) tiles in f16
    __shared__ _Float16 Ks[2][BC * KST];
    __shared__ _Float16 VT[2][DH * KST];

    const int tid  = threadIdx.x;
    const int wave = tid >> 5;
    const int lane = tid & 31;
    const int half = lane >> 4;     // 0 | 1 : selects K-chunk offset per WMMA A/B layout
    const int l16  = lane & 15;

    const int H  = gridDim.y;
    const size_t bh = (size_t)blockIdx.z * H + blockIdx.y;
    const float* Qb = Qg + bh * (size_t)SEQ_L * DH;
    const float* Kb = Kg + bh * (size_t)SEQ_L * DH;
    const float* Vb = Vg + bh * (size_t)SEQ_L * DH;
    const int*   Mb = Mg + (size_t)blockIdx.z * SEQ_L * SEQ_L;
    float*       Ob = Og + bh * (size_t)SEQ_L * DH;

    // Query row this lane owns (as a column of S^T / O^T)
    const int qrow = blockIdx.x * BR + wave * 16 + l16;

    // ---- Q as B-operand fragments (contraction over d), prescaled by 1/sqrt(64) ----
    H16 qf0, qf1;
    {
        const float* qp = Qb + (size_t)qrow * DH + half * 8;
        f4v a0 = *(const f4v*)(qp +  0), a1 = *(const f4v*)(qp +  4);
        f4v b0 = *(const f4v*)(qp + 16), b1 = *(const f4v*)(qp + 20);
        f4v c0 = *(const f4v*)(qp + 32), c1 = *(const f4v*)(qp + 36);
        f4v d0 = *(const f4v*)(qp + 48), d1 = *(const f4v*)(qp + 52);
        #pragma unroll
        for (int i = 0; i < 4; ++i) {
            qf0.e[i]      = (_Float16)(a0[i] * 0.125f);
            qf0.e[4 + i]  = (_Float16)(a1[i] * 0.125f);
            qf0.e[8 + i]  = (_Float16)(b0[i] * 0.125f);
            qf0.e[12 + i] = (_Float16)(b1[i] * 0.125f);
            qf1.e[i]      = (_Float16)(c0[i] * 0.125f);
            qf1.e[4 + i]  = (_Float16)(c1[i] * 0.125f);
            qf1.e[8 + i]  = (_Float16)(d0[i] * 0.125f);
            qf1.e[12 + i] = (_Float16)(d1[i] * 0.125f);
        }
    }

    // cooperative staging coordinates (fixed per thread)
    const int sr = tid >> 2;            // key row 0..63
    const int sc = (tid & 3) << 4;      // d column 0,16,32,48

    // in-flight K/V tile (global f32 loads for the NEXT block)
    f4v kr0, kr1, kr2, kr3, vr0, vr1, vr2, vr3;

    auto gload = [&](int kb) {
        const float* kp = Kb + (size_t)(kb + sr) * DH + sc;
        kr0 = *(const f4v*)(kp + 0);  kr1 = *(const f4v*)(kp + 4);
        kr2 = *(const f4v*)(kp + 8);  kr3 = *(const f4v*)(kp + 12);
        const float* vp = Vb + (size_t)(kb + sr) * DH + sc;
        vr0 = *(const f4v*)(vp + 0);  vr1 = *(const f4v*)(vp + 4);
        vr2 = *(const f4v*)(vp + 8);  vr3 = *(const f4v*)(vp + 12);
    };

    auto lstore = [&](int p) {
        H8 p0, p1;
        #pragma unroll
        for (int i = 0; i < 4; ++i) {
            p0.e[i]     = (_Float16)kr0[i];
            p0.e[4 + i] = (_Float16)kr1[i];
            p1.e[i]     = (_Float16)kr2[i];
            p1.e[4 + i] = (_Float16)kr3[i];
        }
        *(h8v*)&Ks[p][sr * KST + sc]     = p0.v;
        *(h8v*)&Ks[p][sr * KST + sc + 8] = p1.v;
        #pragma unroll
        for (int i = 0; i < 4; ++i) {
            VT[p][(sc + i) * KST + sr]      = (_Float16)vr0[i];
            VT[p][(sc + 4 + i) * KST + sr]  = (_Float16)vr1[i];
            VT[p][(sc + 8 + i) * KST + sr]  = (_Float16)vr2[i];
            VT[p][(sc + 12 + i) * KST + sr] = (_Float16)vr3[i];
        }
    };

    // Online-softmax state (per query column; lanes l and l+16 stay consistent via shfl)
    float m_run = -1e30f;
    float l_run = 0.0f;
    f8v acc[4];                       // O^T accumulators: 4 d-tiles of 16
    #pragma unroll
    for (int t = 0; t < 4; ++t)
        #pragma unroll
        for (int r = 0; r < 8; ++r) acc[t][r] = 0.0f;

    // prologue: stage block 0
    gload(0);
    lstore(0);

    for (int it = 0; it < NBLK; ++it) {
        const int kb = it * BC;
        const int p  = it & 1;

        // one barrier per block: separates prior block's LDS reads from this
        // block's writes, and makes buffer p's contents visible
        __syncthreads();

        // kick off global loads for the next block (hidden behind WMMA work)
        if (it + 1 < NBLK) gload(kb + BC);

        // ---- S^T tiles: S^T = K * Q^T  (4 tiles of 16 keys x 16 queries) ----
        f8v st[4];
        #pragma unroll
        for (int kt = 0; kt < 4; ++kt) {
            const _Float16* kr = &Ks[p][(kt * 16 + l16) * KST + half * 8];
            H16 a0, a1;
            a0.h[0] = *(const h8v*)(kr + 0);
            a0.h[1] = *(const h8v*)(kr + 16);
            a1.h[0] = *(const h8v*)(kr + 32);
            a1.h[1] = *(const h8v*)(kr + 48);
            f8v c = {};
            c = __builtin_amdgcn_wmma_f32_16x16x32_f16(false, a0.v, false, qf0.v,
                                                       (short)0, c, false, false);
            c = __builtin_amdgcn_wmma_f32_16x16x32_f16(false, a1.v, false, qf1.v,
                                                       (short)0, c, false, false);
            // mask: element r holds key kb + kt*16 + half*8 + r for query qrow
            const int* mp = Mb + (size_t)qrow * SEQ_L + kb + kt * 16 + half * 8;
            i4v m0 = *(const i4v*)mp;
            i4v m1 = *(const i4v*)(mp + 4);
            #pragma unroll
            for (int r = 0; r < 4; ++r) {
                c[r]     = (m0[r] == 0) ? -1e9f : c[r];
                c[4 + r] = (m1[r] == 0) ? -1e9f : c[4 + r];
            }
            st[kt] = c;
        }

        // ---- online softmax over this 64-key block ----
        float bmax = -1e30f;
        #pragma unroll
        for (int kt = 0; kt < 4; ++kt)
            #pragma unroll
            for (int r = 0; r < 8; ++r) bmax = fmaxf(bmax, st[kt][r]);
        bmax = fmaxf(bmax, __shfl_xor(bmax, 16, 32));   // merge complementary key halves
        const float mnew  = fmaxf(m_run, bmax);
        const float alpha = __expf(m_run - mnew);

        H16 pb0, pb1;            // P^T as f16 B-fragments: keys 0..31 and 32..63
        float bsum = 0.0f;
        #pragma unroll
        for (int kt = 0; kt < 4; ++kt) {
            #pragma unroll
            for (int r = 0; r < 8; ++r) {
                float pv = __expf(st[kt][r] - mnew);
                bsum += pv;
                if (kt < 2) pb0.e[(kt & 1) * 8 + r] = (_Float16)pv;
                else        pb1.e[(kt & 1) * 8 + r] = (_Float16)pv;
            }
        }
        bsum += __shfl_xor(bsum, 16, 32);
        l_run = l_run * alpha + bsum;
        m_run = mnew;

        #pragma unroll
        for (int t = 0; t < 4; ++t)
            #pragma unroll
            for (int r = 0; r < 8; ++r) acc[t][r] *= alpha;

        // ---- O^T += V^T * P^T  (4 d-tiles, contraction over 64 keys) ----
        #pragma unroll
        for (int dt = 0; dt < 4; ++dt) {
            const _Float16* vr = &VT[p][(dt * 16 + l16) * KST + half * 8];
            H16 a0, a1;
            a0.h[0] = *(const h8v*)(vr + 0);    // keys  0..31 chunk
            a0.h[1] = *(const h8v*)(vr + 16);
            a1.h[0] = *(const h8v*)(vr + 32);   // keys 32..63 chunk
            a1.h[1] = *(const h8v*)(vr + 48);
            acc[dt] = __builtin_amdgcn_wmma_f32_16x16x32_f16(false, a0.v, false, pb0.v,
                                                             (short)0, acc[dt], false, false);
            acc[dt] = __builtin_amdgcn_wmma_f32_16x16x32_f16(false, a1.v, false, pb1.v,
                                                             (short)0, acc[dt], false, false);
        }

        // drain the prefetched tile into the alternate buffer (read in it+1)
        if (it + 1 < NBLK) lstore(p ^ 1);
    }

    // ---- normalize and store O (fp32). lane holds query qrow, d = dt*16 + half*8 + r ----
    const float inv = 1.0f / l_run;
    #pragma unroll
    for (int dt = 0; dt < 4; ++dt) {
        float* op = Ob + (size_t)qrow * DH + dt * 16 + half * 8;
        f4v o0, o1;
        #pragma unroll
        for (int r = 0; r < 4; ++r) {
            o0[r] = acc[dt][r] * inv;
            o1[r] = acc[dt][4 + r] * inv;
        }
        *(f4v*)(op + 0) = o0;
        *(f4v*)(op + 4) = o1;
    }
}

extern "C" void kernel_launch(void* const* d_in, const int* in_sizes, int n_in,
                              void* d_out, int out_size, void* d_ws, size_t ws_size,
                              hipStream_t stream) {
    (void)n_in; (void)out_size; (void)d_ws; (void)ws_size;
    const float* q = (const float*)d_in[0];
    const float* k = (const float*)d_in[1];
    const float* v = (const float*)d_in[2];
    const int*   m = (const int*)d_in[3];
    float* out = (float*)d_out;

    const int b = in_sizes[3] / (SEQ_L * SEQ_L);                    // mask is b*1*L*L
    const int h = (int)((size_t)in_sizes[0] / ((size_t)b * SEQ_L * DH));

    dim3 grid(SEQ_L / BR, h, b);   // (16, 16, 2)
    fattn_kernel<<<grid, 256, 0, stream>>>(q, k, v, m, out);
}
